// PTLSTM_52621939310724
// MI455X (gfx1250) — compile-verified, matrix operand order checked by reading, not run on hardware
//
#include <hip/hip_runtime.h>
#include <hip/hip_bf16.h>

typedef __bf16 bf16_t;
typedef __attribute__((ext_vector_type(16))) __bf16 v16bf;
typedef __attribute__((ext_vector_type(8)))  __bf16 v8bf;
typedef __attribute__((ext_vector_type(4)))  __bf16 v4bf;
typedef __attribute__((ext_vector_type(8)))  float  v8f;
typedef __attribute__((ext_vector_type(4)))  unsigned int u32x4;
typedef __attribute__((ext_vector_type(8)))  int  i32x8;
typedef __attribute__((ext_vector_type(4)))  int  i32x4;

#define NB   32          // batch
#define TT   2048        // timesteps
#define DD   256         // input dim
#define HH   256         // hidden dim
#define G4   1024        // 4*H
#define HPAD 264         // padded LDS row stride (bf16 elems), 528B -> bank-quad spread

// Load a 16x32 bf16 WMMA operand fragment (A or B role).
// p points at element [row(lane)][kbase + half*8]; layout per CDNA5 ISA:
//   elems 0..7  -> K = kbase + half*8 + e
//   elems 8..15 -> K = kbase + 16 + half*8 + (e-8)
__device__ inline v16bf ldfrag(const bf16_t* p) {
    v8bf lo = *(const v8bf*)p;
    v8bf hi = *(const v8bf*)(p + 16);
    v16bf r;
#pragma unroll
    for (int i = 0; i < 8; ++i) { r[i] = lo[i]; r[i + 8] = hi[i]; }
    return r;
}

__device__ inline v8f wmma_bf16(v16bf a, v16bf b, v8f c) {
    return __builtin_amdgcn_wmma_f32_16x16x32_bf16(false, a, false, b,
                                                   (short)0, c, false, false);
}

__device__ inline float sigf(float x) {
    return 1.0f / (1.0f + __expf(-x));
}
__device__ inline float tanh_fast(float x) {
    float ax = fabsf(x);
    float e  = __expf(-2.0f * ax);          // in (0,1], no overflow
    float r  = (1.0f - e) / (1.0f + e);
    return copysignf(r, x);
}

// ---- Tensor Data Mover: contiguous 128KB (32768 dwords) global -> LDS -----
// 1-D tile: tensor_dim0 = tile_dim0 = 32768 dwords, data_size = 4B.
__device__ __forceinline__ void tdm_load_xg_slice(const float* gsrc, void* lds_dst) {
    unsigned lds_off = (unsigned)(size_t)lds_dst;              // low 32b = LDS offset
    unsigned long long ga = (unsigned long long)(size_t)gsrc;
    const unsigned NW = NB * G4;                               // 32768 dwords

    u32x4 g0;
    g0[0] = 1u;                                                // count=1, load, user D#
    g0[1] = lds_off;                                           // lds_addr (bytes)
    g0[2] = (unsigned)(ga & 0xFFFFFFFFu);                      // global_addr[31:0]
    g0[3] = (unsigned)((ga >> 32) & 0x01FFFFFFu) | (2u << 30); // addr[56:32] | type=2

    i32x8 g1;
    g1[0] = (int)(2u << 16);                                   // data_size = 4B (code 2)
    g1[1] = (int)((NW & 0xFFFFu) << 16);                       // tensor_dim0[15:0]
    g1[2] = (int)(((NW >> 16) & 0xFFFFu) | (1u << 16));        // dim0[31:16], tensor_dim1=1
    g1[3] = (int)((NW & 0xFFFFu) << 16);                       // tile_dim0 = NW
    g1[4] = 0;                                                 // tile_dim1/2 = 0 (1-D)
    g1[5] = (int)NW;                                           // tensor_dim0_stride[31:0]
    g1[6] = 0;
    g1[7] = 0;

    i32x4 zz = {0, 0, 0, 0};
#if defined(__clang_major__) && (__clang_major__ >= 23)
    i32x8 z8 = {0, 0, 0, 0, 0, 0, 0, 0};
    __builtin_amdgcn_tensor_load_to_lds(g0, g1, zz, zz, z8, 0);
#else
    __builtin_amdgcn_tensor_load_to_lds(g0, g1, zz, zz, 0);
#endif
}

// ---- conversion kernels ---------------------------------------------------
__global__ __launch_bounds__(256) void cvt_weights(
        const float* __restrict__ wih, const float* __restrict__ whh,
        const float* __restrict__ bih, const float* __restrict__ bhh,
        bf16_t* __restrict__ wihb, bf16_t* __restrict__ whhb,
        float* __restrict__ bias) {
    int i = blockIdx.x * 256 + threadIdx.x;       // grid covers 1024*256
    wihb[i] = (bf16_t)wih[i];
    whhb[i] = (bf16_t)whh[i];
    if (i < G4) bias[i] = bih[i] + bhh[i];
}

__global__ __launch_bounds__(256) void cvt_x(const float* __restrict__ x,
                                             bf16_t* __restrict__ xb) {
    size_t i = ((size_t)blockIdx.x * 256 + threadIdx.x) * 4;
    float4 v = *(const float4*)(x + i);
    v4bf o;
    o[0] = (bf16_t)v.x; o[1] = (bf16_t)v.y; o[2] = (bf16_t)v.z; o[3] = (bf16_t)v.w;
    *(v4bf*)(xb + i) = o;
}

// ---- phase 1: xg[t][n][g] = x[n][t][:] . w_ih[g][:] + bias[g] -------------
// grid: 32768 blocks of 256 (8 waves). block -> (n, t_tile, ngroup); wave -> col tile.
__global__ __launch_bounds__(256) void gemm_xg(
        const bf16_t* __restrict__ xb,   // (N*T, D) bf16 row-major
        const bf16_t* __restrict__ wb,   // (4H, D)  bf16 row-major (acts as B^T)
        const float*  __restrict__ bias, // (4H)
        float* __restrict__ xg) {        // (T, N, 4H) f32  -- time-major!
    int lane = threadIdx.x & 31;
    int wv   = threadIdx.x >> 5;
    int lanelo = lane & 15, half = lane >> 4;

    int bid = blockIdx.x;
    int ngroup = bid & 7;
    int mt = bid >> 3;              // 0..4095
    int n  = mt >> 7;               // batch 0..31
    int tt = mt & 127;              // t-tile 0..127
    int ct = ngroup * 8 + wv;       // gate-col tile 0..63
    int gcol = ct * 16 + lanelo;

    const bf16_t* ap = xb + (size_t)(n * TT + tt * 16 + lanelo) * DD + half * 8;
    const bf16_t* bp = wb + (size_t)gcol * DD + half * 8;

    v8f acc = {};
#pragma unroll
    for (int k = 0; k < 8; ++k) {
        v16bf a = ldfrag(ap + k * 32);
        v16bf b = ldfrag(bp + k * 32);
        acc = wmma_bf16(a, b, acc);
    }
    float bv = bias[gcol];
#pragma unroll
    for (int e = 0; e < 8; ++e) {
        int trow = tt * 16 + e + 8 * half;               // time index
        xg[((size_t)trow * NB + n) * G4 + gcol] = acc[e] + bv;
    }
}

// ---- phase 2: sequential LSTM scan, one persistent workgroup --------------
// 1024 threads = 32 waves. wave -> (m_tile in {0,1}, unit_tile in 0..15).
// h: LDS bf16 double-buffered.  c: registers.  xg_t: LDS f32 double-buffered,
// staged one step ahead by the Tensor Data Mover (TENSORcnt).
__global__ __launch_bounds__(1024) void lstm_scan(
        const float*  __restrict__ xg,    // (T, N, 4H) f32
        const bf16_t* __restrict__ whh,   // (4H, H) bf16 row-major
        const float*  __restrict__ state, // (N, 2H) f32
        float* __restrict__ out) {        // output (N,T,H) ++ outstate (N,2H)
    __shared__ float  xs[2][NB][G4];      // 2 x 128 KB staging for xg_t
    __shared__ bf16_t hs[2][NB][HPAD];    // 2 x ~17 KB for h

    int tid  = threadIdx.x;
    int lane = tid & 31, wv = tid >> 5;
    int lanelo = lane & 15, half = lane >> 4;
    int mt = wv & 1;              // batch tile
    int ut = wv >> 1;             // hidden-unit tile
    int mbase = mt * 16;
    int ucol  = ut * 16;

    // kick off TDM for step 0's xg slice immediately
    if (wv == 0) tdm_load_xg_slice(xg, &xs[0][0][0]);

    // h0 -> LDS buffer 0 (bf16)
    for (int i = tid; i < NB * HH; i += 1024) {
        int r = i >> 8, cidx = i & 255;
        hs[0][r][cidx] = (bf16_t)state[r * (2 * HH) + cidx];
    }
    // c0 fragment in registers, matching the WMMA C/D lane layout
    v8f c;
#pragma unroll
    for (int e = 0; e < 8; ++e)
        c[e] = state[(mbase + e + 8 * half) * (2 * HH) + HH + ucol + lanelo];

    if (wv == 0) __builtin_amdgcn_s_wait_tensorcnt(0);
    __syncthreads();

    const bf16_t* bpB = whh + (size_t)(ucol + lanelo) * HH + half * 8;
    float* os = out + (size_t)NB * TT * HH;   // outstate region

    for (int t = 0; t < TT; ++t) {
        int cur = t & 1, nxt = cur ^ 1;

        // async-stage next step's xg while this step computes
        if (wv == 0 && t + 1 < TT)
            tdm_load_xg_slice(xg + (size_t)(t + 1) * NB * G4, &xs[nxt][0][0]);

        // gates = h @ w_hh^T : K=256 in 8 steps, A shared across 4 gate WMMAs
        v8f ai = {}, af = {}, ag = {}, ao = {};
#pragma unroll
        for (int k = 0; k < 8; ++k) {
            int kb = k * 32 + half * 8;
            v16bf a  = ldfrag(&hs[cur][mbase + lanelo][kb]);   // ds_load_b128 x2
            v16bf b0 = ldfrag(bpB + kb);                       // gate i rows
            v16bf b1 = ldfrag(bpB + (size_t)1 * HH * HH + kb); // gate f rows
            v16bf b2 = ldfrag(bpB + (size_t)2 * HH * HH + kb); // gate g rows
            v16bf b3 = ldfrag(bpB + (size_t)3 * HH * HH + kb); // gate o rows
            ai = wmma_bf16(a, b0, ai);
            af = wmma_bf16(a, b1, af);
            ag = wmma_bf16(a, b2, ag);
            ao = wmma_bf16(a, b3, ao);
        }

        // add xg (from LDS, TDM-written; volatile so loads aren't folded),
        // gate math; c stays in registers, h -> LDS(next) + output
        const volatile float* xcur = &xs[cur][0][0];
#pragma unroll
        for (int e = 0; e < 8; ++e) {
            int n = mbase + e + 8 * half;
            int u = ucol + lanelo;
            const volatile float* rp = xcur + (size_t)n * G4 + u;
            float iv = sigf(ai[e] + rp[0]);
            float fv = sigf(af[e] + rp[HH]);
            float gv = tanh_fast(ag[e] + rp[2 * HH]);
            float ov = sigf(ao[e] + rp[3 * HH]);
            float cn = fv * c[e] + iv * gv;
            c[e] = cn;
            float hv = ov * tanh_fast(cn);
            hs[nxt][n][u] = (bf16_t)hv;
            out[((size_t)n * TT + t) * HH + u] = hv;
        }

        // TDM for t+1 must land before anyone reads xs[nxt] next step
        if (wv == 0 && t + 1 < TT) __builtin_amdgcn_s_wait_tensorcnt(0);
        __syncthreads();
    }

    // final state: h from last LDS buffer (t=2048 -> buffer 0), c from registers
#pragma unroll
    for (int e = 0; e < 8; ++e) {
        int n = mbase + e + 8 * half;
        int u = ucol + lanelo;
        os[n * (2 * HH) + u]      = (float)hs[0][n][u];
        os[n * (2 * HH) + HH + u] = c[e];
    }
}

extern "C" void kernel_launch(void* const* d_in, const int* in_sizes, int n_in,
                              void* d_out, int out_size, void* d_ws, size_t ws_size,
                              hipStream_t stream) {
    const float* x     = (const float*)d_in[0];  // (32,2048,256)
    const float* state = (const float*)d_in[1];  // (32,512)
    const float* w_ih  = (const float*)d_in[2];  // (1024,256)
    const float* w_hh  = (const float*)d_in[3];  // (1024,256)
    const float* b_ih  = (const float*)d_in[4];  // (1024)
    const float* b_hh  = (const float*)d_in[5];  // (1024)
    float* out = (float*)d_out;

    // workspace layout
    char* ws = (char*)d_ws;
    float*  xg    = (float*)ws;                                    // 256 MB
    bf16_t* xbf   = (bf16_t*)(ws + (size_t)TT * NB * G4 * 4);      // 32 MB
    bf16_t* wihb  = (bf16_t*)((char*)xbf + (size_t)NB * TT * DD * 2);
    bf16_t* whhb  = (bf16_t*)((char*)wihb + (size_t)G4 * DD * 2);
    float*  bias  = (float*)((char*)whhb + (size_t)G4 * HH * 2);

    cvt_weights<<<(G4 * DD) / 256, 256, 0, stream>>>(w_ih, w_hh, b_ih, b_hh,
                                                     wihb, whhb, bias);
    cvt_x<<<(NB * TT * DD) / (256 * 4), 256, 0, stream>>>(x, xbf);
    gemm_xg<<<(NB * (TT / 16)) * 8, 256, 0, stream>>>(xbf, wihb, bias, xg);
    lstm_scan<<<1, 1024, 0, stream>>>(xg, whhb, state, out);
}
